// ConvFormer_12773232738766
// MI455X (gfx1250) — compile-verified, matrix operand order checked by reading, not run on hardware
//
#include <hip/hip_runtime.h>

typedef __attribute__((ext_vector_type(16))) _Float16 v16h;
typedef __attribute__((ext_vector_type(8)))  float    v8f;

#define NHEADS 8
#define KQV    64
#define C_IN   256
#define BATCH  8
#define SLEN   1024
#define BS     (BATCH * SLEN)       // 8192 rows for the output projection
#define PROJ_K (NHEADS * KQV)       // 512

// ---------------------------------------------------------------------------
// WMMA helper (CDNA5 wave32, 16x16x32 f16 -> f32 accumulate)
// ---------------------------------------------------------------------------
static __device__ __forceinline__ v8f wmma_f16(v16h a, v16h b, v8f c) {
  // (neg_a, A, neg_b, B, c_mod, C, reuse_a, reuse_b)
  return __builtin_amdgcn_wmma_f32_16x16x32_f16(false, a, false, b, (short)0, c,
                                                false, false);
}

// Load a 16x32 f16 fragment from a row-major tile (16 rows, >=32 cols, stride ld).
// CDNA5 16-bit A layout: lanes 0-15 hold M=lane; VGPR e/2 holds K pairs.
//   element e -> K = (e<8 ? 0 : 16) + (lane>=16 ? 8 : 0) + (e&7)
// The B operand uses the mirrored mapping (lane -> N), so supplying B as
// B^T row-major ([N rows][K cols]) uses the identical loader.
static __device__ __forceinline__ v16h frag_h(const _Float16* p, int ld) {
  const int l  = threadIdx.x & 31;
  const int r  = l & 15;
  const int hi = l >> 4;
  const _Float16* row = p + (size_t)r * ld;
  v16h f{};
#pragma unroll
  for (int e = 0; e < 16; ++e) {
    const int k = ((e < 8) ? 0 : 16) + hi * 8 + (e & 7);
    f[e] = row[k];
  }
  return f;
}

// ---------------------------------------------------------------------------
// Kernel 1: xp = x + PE(c,h,w); write xp (f32, [B,C,S]) and xpT (f16, [B,S,C])
// ---------------------------------------------------------------------------
__global__ void pe_add_kernel(const float* __restrict__ x,
                              float* __restrict__ xp,
                              _Float16* __restrict__ xpT) {
  const int idx = blockIdx.x * 256 + threadIdx.x;   // < B*C*S
  const int s  = idx & (SLEN - 1);
  const int bc = idx >> 10;
  const int c  = bc & (C_IN - 1);
  const int b  = bc >> 8;
  const int hh = s >> 5;
  const int ww = s & 31;
  // div_term[j] = exp(-(ln 10000 / 128) * 2j) = exp(-0.14391157 * j)
  float pe;
  if (c < 128) {
    const int j = c >> 1;
    const float a = (float)ww * __expf(-0.14391156919f * (float)j);
    pe = (c & 1) ? __cosf(a) : __sinf(a);
  } else {
    const int j = (c - 128) >> 1;
    const float a = (float)hh * __expf(-0.14391156919f * (float)j);
    pe = (c & 1) ? __cosf(a) : __sinf(a);
  }
  const float v = x[idx] + pe;
  xp[idx] = v;
  xpT[((size_t)b * SLEN + s) * C_IN + c] = (_Float16)v;
}

// ---------------------------------------------------------------------------
// Kernel 2: one-shot f32 -> f16 conversion of all weight matrices.
//   Wk/Wq/Wv: [N, KQV, C] (131072 elems each); Wl: [C, 512] (131072 elems).
//   Converted once; afterwards the GEMMs do pure f16 b128 loads (L2-resident).
// ---------------------------------------------------------------------------
__global__ void wcvt_kernel(const float* __restrict__ Wk, const float* __restrict__ Wq,
                            const float* __restrict__ Wv, const float* __restrict__ Wl,
                            _Float16* __restrict__ Wk16, _Float16* __restrict__ Wq16,
                            _Float16* __restrict__ Wv16, _Float16* __restrict__ Wl16) {
  const int idx   = blockIdx.x * 256 + threadIdx.x;   // < 4 * 131072
  const int which = idx >> 17;
  const int i     = idx & ((1 << 17) - 1);
  const float* src = (which == 0) ? Wk : (which == 1) ? Wq : (which == 2) ? Wv : Wl;
  _Float16*    dst = (which == 0) ? Wk16 : (which == 1) ? Wq16 : (which == 2) ? Wv16 : Wl16;
  dst[i] = (_Float16)src[i];
}

// ---------------------------------------------------------------------------
// Kernel 3: QKV projections. One wave computes a 16x64 output block
//   (full head dim), sharing each A-fragment across 4 WMMAs.
//   K,Q stored [b,n,s,64] f16; V stored transposed [b,n,kq,S].
// grid = (S/16, B*N*3), block = 32
// ---------------------------------------------------------------------------
__global__ void qkv_kernel(const _Float16* __restrict__ xpT,
                           const _Float16* __restrict__ Wk16, const float* __restrict__ bk,
                           const _Float16* __restrict__ Wq16, const float* __restrict__ bq,
                           const _Float16* __restrict__ Wv16, const float* __restrict__ bv,
                           _Float16* __restrict__ Kp,
                           _Float16* __restrict__ Qp,
                           _Float16* __restrict__ Vt) {
  const int z     = blockIdx.y;
  const int which = z % 3;            // 0=K, 1=Q, 2=V
  const int bn    = z / 3;
  const int n     = bn & (NHEADS - 1);
  const int b     = bn >> 3;
  const int s0    = blockIdx.x * 16;

  const _Float16* W  = (which == 0 ? Wk16 : which == 1 ? Wq16 : Wv16) + (size_t)n * KQV * C_IN;
  const float* bias  = (which == 0 ? bk : which == 1 ? bq : bv) + n * KQV;

  const _Float16* Arow = xpT + ((size_t)b * SLEN + s0) * C_IN;
  v8f acc[4] = {v8f{}, v8f{}, v8f{}, v8f{}};
#pragma unroll
  for (int c0 = 0; c0 < C_IN; c0 += 32) {
    const v16h a = frag_h(Arow + c0, C_IN);                  // A: xpT[s, c]
#pragma unroll
    for (int j = 0; j < 4; ++j) {                            // B^T: W[kq, c]
      const v16h bb = frag_h(W + (size_t)(j * 16) * C_IN + c0, C_IN);
      acc[j] = wmma_f16(a, bb, acc[j]);
    }
  }

  const int l  = threadIdx.x & 31;
  const int nn = l & 15;
  const int hi = l >> 4;
#pragma unroll
  for (int j = 0; j < 4; ++j) {
    const int k = j * 16 + nn;
    const float bsc = bias[k];
#pragma unroll
    for (int r = 0; r < 8; ++r) {
      const int s = s0 + r + 8 * hi;
      const _Float16 hval = (_Float16)(acc[j][r] + bsc);
      if (which == 2) {
        Vt[(((size_t)b * NHEADS + n) * KQV + k) * SLEN + s] = hval;  // [b,n,kq,S]
      } else {
        _Float16* dst = (which == 0) ? Kp : Qp;
        dst[(((size_t)b * NHEADS + n) * SLEN + s) * KQV + k] = hval; // [b,n,s,64]
      }
    }
  }
}

// ---------------------------------------------------------------------------
// Kernel 4: fused attention with cross-HEAD softmax.
//   Workgroup = 8 waves (one head each) handling one 16-row s-tile of batch b.
//   Per 32-wide t-tile: scores via WMMA -> LDS -> softmax over the 8 heads
//   -> sm (f16) as A fragment -> WMMA accumulate with V.
//   The [B,N,S,S] tensor never touches HBM; per-batch K/Q/V (~3 MB) stays in L2.
// grid = (S/16, B), block = 256
// ---------------------------------------------------------------------------
__global__ void attn_kernel(const _Float16* __restrict__ Kp,
                            const _Float16* __restrict__ Qp,
                            const _Float16* __restrict__ Vt,
                            _Float16* __restrict__ ATT) {
  __shared__ float    Pls[NHEADS][16 * 32];   // 16 KB: raw scores (all heads)
  __shared__ _Float16 SMls[NHEADS][16 * 32];  //  8 KB: softmaxed, own head only

  const int s0 = blockIdx.x * 16;
  const int b  = blockIdx.y;
  const int n  = threadIdx.x >> 5;            // head = wave id
  const int l  = threadIdx.x & 31;
  const int nn = l & 15;
  const int hi = l >> 4;

  const _Float16* Kh = Kp + ((size_t)b * NHEADS + n) * SLEN * KQV;
  const _Float16* Qh = Qp + ((size_t)b * NHEADS + n) * SLEN * KQV;
  const _Float16* Vh = Vt + ((size_t)b * NHEADS + n) * KQV * SLEN;

  // K rows for this s-tile stay resident: 16x64 = two A fragments
  const v16h aK0 = frag_h(Kh + (size_t)s0 * KQV + 0,  KQV);
  const v16h aK1 = frag_h(Kh + (size_t)s0 * KQV + 32, KQV);

  v8f acc[4] = {v8f{}, v8f{}, v8f{}, v8f{}};  // att[s-tile, 0:64]
  const float invS = 1.0f / (float)SLEN;

  for (int t0 = 0; t0 < SLEN; t0 += 32) {
    if (t0 + 32 < SLEN) __builtin_prefetch(Qh + (size_t)(t0 + 32) * KQV, 0, 1);

    // scores P[s, t0:t0+32] = K[s,:] . Q[t,:]  (B supplied as Q row-major)
    v8f p0{}, p1{};
    p0 = wmma_f16(aK0, frag_h(Qh + (size_t)t0 * KQV + 0,  KQV), p0);
    p0 = wmma_f16(aK1, frag_h(Qh + (size_t)t0 * KQV + 32, KQV), p0);
    p1 = wmma_f16(aK0, frag_h(Qh + (size_t)(t0 + 16) * KQV + 0,  KQV), p1);
    p1 = wmma_f16(aK1, frag_h(Qh + (size_t)(t0 + 16) * KQV + 32, KQV), p1);

    __syncthreads();  // previous iteration's cross-head P reads are done
#pragma unroll
    for (int r = 0; r < 8; ++r) {
      const int row = (r + 8 * hi) * 32;
      Pls[n][row + nn]      = p0[r] * invS;
      Pls[n][row + 16 + nn] = p1[r] * invS;
    }
    __syncthreads();  // all heads' P visible

    // softmax over the head axis for the 16 elements this lane owns
#pragma unroll
    for (int r = 0; r < 8; ++r) {
#pragma unroll
      for (int half = 0; half < 2; ++half) {
        const int idx = (r + 8 * hi) * 32 + half * 16 + nn;
        float vh[NHEADS];
        float mx = -3.4e38f;
#pragma unroll
        for (int h = 0; h < NHEADS; ++h) { vh[h] = Pls[h][idx]; mx = fmaxf(mx, vh[h]); }
        float sum = 0.0f, mine = 0.0f;
#pragma unroll
        for (int h = 0; h < NHEADS; ++h) {
          const float e = __expf(vh[h] - mx);
          sum += e;
          if (h == n) mine = e;
        }
        SMls[n][idx] = (_Float16)(mine / sum);
      }
    }
    // own-head LDS round trip re-lays sm into the A-fragment pattern
    const v16h aSM = frag_h(&SMls[n][0], 32);
#pragma unroll
    for (int j = 0; j < 4; ++j) {
      // B^T row-major: Vt[kq, t] rows (N=kq chunk), cols t0:t0+32
      const v16h bV = frag_h(Vh + (size_t)(j * 16) * SLEN + t0, SLEN);
      acc[j] = wmma_f16(aSM, bV, acc[j]);
    }
  }

  // store att in the projection layout [B, S, kq*8 + n] (kq slow, head fast)
#pragma unroll
  for (int j = 0; j < 4; ++j) {
#pragma unroll
    for (int r = 0; r < 8; ++r) {
      const int s  = s0 + r + 8 * hi;
      const int kq = j * 16 + nn;
      ATT[((size_t)b * SLEN + s) * PROJ_K + kq * NHEADS + n] = (_Float16)acc[j][r];
    }
  }
}

// ---------------------------------------------------------------------------
// Kernel 5: output projection + leaky-relu + residual -> y (pre-BN) in d_out.
//   One wave computes a 16x64 block (4 c-tiles), sharing each A-fragment.
// grid = (BS/16, C/64), block = 32
// ---------------------------------------------------------------------------
__global__ void proj_kernel(const _Float16* __restrict__ ATT,
                            const _Float16* __restrict__ Wl16,
                            const float* __restrict__ bl,
                            const float* __restrict__ xp,
                            float* __restrict__ out) {
  const int row0 = blockIdx.x * 16;   // over B*S
  const int c0   = blockIdx.y * 64;
  v8f acc[4] = {v8f{}, v8f{}, v8f{}, v8f{}};
#pragma unroll
  for (int col0 = 0; col0 < PROJ_K; col0 += 32) {
    const v16h a = frag_h(ATT + (size_t)row0 * PROJ_K + col0, PROJ_K);
#pragma unroll
    for (int j = 0; j < 4; ++j) {                 // B^T: Wl[c, col]
      const v16h bb = frag_h(Wl16 + (size_t)(c0 + j * 16) * PROJ_K + col0, PROJ_K);
      acc[j] = wmma_f16(a, bb, acc[j]);
    }
  }
  const int l  = threadIdx.x & 31;
  const int nn = l & 15;
  const int hi = l >> 4;
#pragma unroll
  for (int j = 0; j < 4; ++j) {
    const int c = c0 + j * 16 + nn;
    const float bias = bl[c];
#pragma unroll
    for (int r = 0; r < 8; ++r) {
      const int row = row0 + r + 8 * hi;
      const int b   = row >> 10;
      const int s   = row & (SLEN - 1);
      float v = acc[j][r] + bias;
      v = (v >= 0.0f) ? v : 0.1f * v;             // LeakyReLU(0.1)
      const size_t oidx = ((size_t)b * C_IN + c) * SLEN + s;
      out[oidx] = v + xp[oidx];                   // residual with xp
    }
  }
}

// ---------------------------------------------------------------------------
// Kernel 6: BatchNorm statistics per channel (biased var) -> scale/shift.
// grid = C, block = 256
// ---------------------------------------------------------------------------
__global__ void bn_stats_kernel(const float* __restrict__ y,
                                const float* __restrict__ gamma,
                                const float* __restrict__ beta,
                                float* __restrict__ stats) {
  const int c   = blockIdx.x;
  const int tid = threadIdx.x;
  float s1 = 0.0f, s2 = 0.0f;
  for (int i = tid; i < BS; i += 256) {
    const int b  = i >> 10;
    const int sp = i & (SLEN - 1);
    const float v = y[((size_t)b * C_IN + c) * SLEN + sp];
    s1 += v;
    s2 += v * v;
  }
  __shared__ float r1[256], r2[256];
  r1[tid] = s1; r2[tid] = s2;
  __syncthreads();
  for (int off = 128; off > 0; off >>= 1) {
    if (tid < off) { r1[tid] += r1[tid + off]; r2[tid] += r2[tid + off]; }
    __syncthreads();
  }
  if (tid == 0) {
    const float mean = r1[0] * (1.0f / (float)BS);
    const float var  = r2[0] * (1.0f / (float)BS) - mean * mean;
    const float sc   = gamma[c] * rsqrtf(var + 1e-5f);
    stats[2 * c]     = sc;
    stats[2 * c + 1] = beta[c] - mean * sc;
  }
}

// ---------------------------------------------------------------------------
// Kernel 7: apply BN in place on d_out (d_out fully rewritten by proj each call)
// ---------------------------------------------------------------------------
__global__ void bn_apply_kernel(float* __restrict__ out,
                                const float* __restrict__ stats) {
  const int idx = blockIdx.x * 256 + threadIdx.x;   // < B*C*S
  const int c = (idx >> 10) & (C_IN - 1);
  out[idx] = out[idx] * stats[2 * c] + stats[2 * c + 1];
}

// ---------------------------------------------------------------------------
extern "C" void kernel_launch(void* const* d_in, const int* in_sizes, int n_in,
                              void* d_out, int out_size, void* d_ws, size_t ws_size,
                              hipStream_t stream) {
  (void)in_sizes; (void)n_in; (void)out_size; (void)ws_size;
  const float* x     = (const float*)d_in[0];
  const float* Wk    = (const float*)d_in[1];
  const float* bk    = (const float*)d_in[2];
  const float* Wq    = (const float*)d_in[3];
  const float* bq    = (const float*)d_in[4];
  const float* Wv    = (const float*)d_in[5];
  const float* bv    = (const float*)d_in[6];
  const float* Wl    = (const float*)d_in[7];
  const float* bl    = (const float*)d_in[8];
  const float* gamma = (const float*)d_in[9];
  const float* beta  = (const float*)d_in[10];
  float* out = (float*)d_out;

  char* p = (char*)d_ws;
  float*    xp   = (float*)p;     p += (size_t)BATCH * C_IN * SLEN * 4;           // 8 MiB
  _Float16* xpT  = (_Float16*)p;  p += (size_t)BATCH * SLEN * C_IN * 2;           // 4 MiB
  _Float16* Kp   = (_Float16*)p;  p += (size_t)BATCH * NHEADS * SLEN * KQV * 2;   // 8 MiB
  _Float16* Qp   = (_Float16*)p;  p += (size_t)BATCH * NHEADS * SLEN * KQV * 2;   // 8 MiB
  _Float16* Vt   = (_Float16*)p;  p += (size_t)BATCH * NHEADS * KQV * SLEN * 2;   // 8 MiB
  _Float16* ATT  = (_Float16*)p;  p += (size_t)BATCH * SLEN * PROJ_K * 2;         // 8 MiB
  _Float16* Wk16 = (_Float16*)p;  p += (size_t)NHEADS * KQV * C_IN * 2;           // 256 KiB
  _Float16* Wq16 = (_Float16*)p;  p += (size_t)NHEADS * KQV * C_IN * 2;           // 256 KiB
  _Float16* Wv16 = (_Float16*)p;  p += (size_t)NHEADS * KQV * C_IN * 2;           // 256 KiB
  _Float16* Wl16 = (_Float16*)p;  p += (size_t)C_IN * PROJ_K * 2;                 // 256 KiB
  float*    stats = (float*)p;

  pe_add_kernel<<<dim3(BATCH * C_IN * SLEN / 256), 256, 0, stream>>>(x, xp, xpT);
  wcvt_kernel<<<dim3(4 * 131072 / 256), 256, 0, stream>>>(Wk, Wq, Wv, Wl,
                                                          Wk16, Wq16, Wv16, Wl16);
  qkv_kernel<<<dim3(SLEN / 16, BATCH * NHEADS * 3), 32, 0, stream>>>(
      xpT, Wk16, bk, Wq16, bq, Wv16, bv, Kp, Qp, Vt);
  attn_kernel<<<dim3(SLEN / 16, BATCH), 256, 0, stream>>>(Kp, Qp, Vt, ATT);
  proj_kernel<<<dim3(BS / 16, C_IN / 64), 32, 0, stream>>>(ATT, Wl16, bl, xp, out);
  bn_stats_kernel<<<dim3(C_IN), 256, 0, stream>>>(out, gamma, beta, stats);
  bn_apply_kernel<<<dim3(BATCH * C_IN * SLEN / 256), 256, 0, stream>>>(out, stats);
}
